// PerformerAttention_43825846288471
// MI455X (gfx1250) — matrix operand order repaired and checked by reading, not compiled, not run on hardware
//
#include <hip/hip_runtime.h>
#include <hip/hip_bf16.h>
#include <math.h>

// ---------------------------------------------------------------------------
// Performer attention for MI455X (gfx1250, wave32, WMMA).
// All GEMMs use v_wmma_f32_16x16x32_bf16 with fp32 accumulation.
// ---------------------------------------------------------------------------

typedef __bf16 bf16;
typedef __attribute__((ext_vector_type(16))) __bf16 v16bf;
typedef __attribute__((ext_vector_type(8)))  __bf16 v8bf;
typedef __attribute__((ext_vector_type(8)))  float  v8f;

union FragBF { v16bf v; v8bf h[2]; };

// A-matrix 16x32 bf16 fragment (ISA 7.12.2): lane holds row M=lane&15;
// elements 0..7  = K[8h .. 8h+7], elements 8..15 = K[16+8h .. 16+8h+7].
__device__ __forceinline__ v16bf lds_a_frag(const bf16* row, int half) {
  FragBF f;
  f.h[0] = *(const v8bf*)(row + 8 * half);
  f.h[1] = *(const v8bf*)(row + 16 + 8 * half);
  return f.v;
}
// B-matrix 32x16 bf16 fragment: lane holds col N=lane&15; elements 0..7 =
// K[8h .. 8h+7], elements 8..15 = K[16+8h .. 16+8h+7] (tile stored [N][K]).
__device__ __forceinline__ v16bf lds_b_frag(const bf16* row, int half) {
  FragBF f;
  f.h[0] = *(const v8bf*)(row + 8 * half);
  f.h[1] = *(const v8bf*)(row + 16 + 8 * half);
  return f.v;
}

#define WMMA_BF16(A, B, C) \
  __builtin_amdgcn_wmma_f32_16x16x32_bf16(false, (A), false, (B), (short)0, (C), false, false)

// ---------------------------------------------------------------------------
// Kernel 1: C[NR,DO] = A[NR,K] @ W[DO,K]^T + bias.  Block tile 128x128, BK=32.
// 8 waves in a 4x2 grid, each wave = 32x64 (2x4 WMMA accumulators).
// W row-major IS the [N][K] layout the B-fragment wants -> no transpose.
// ---------------------------------------------------------------------------
template <bool A_IS_BF16, bool OUT_BF16>
__global__ __launch_bounds__(256)
void gemm_bias_kernel(const void* __restrict__ Ain, const float* __restrict__ W,
                      const float* __restrict__ bias, void* __restrict__ Cout,
                      int K, int DO) {
  __shared__ alignas(16) bf16 lA[128][40];
  __shared__ alignas(16) bf16 lB[128][40];
  const int t = threadIdx.x, lane = t & 31, wv = t >> 5;
  const int wm = wv >> 1, wn = wv & 1;           // 4x2 wave grid
  const int half = lane >> 4, nl = lane & 15;
  const long long rowBase = (long long)blockIdx.y * 128;
  const int colBase = blockIdx.x * 128;

  v8f zero = {};
  v8f acc[2][4];
  for (int i = 0; i < 2; ++i)
    for (int j = 0; j < 4; ++j) acc[i][j] = zero;

  const float* Af = (const float*)Ain;
  const bf16*  Ab = (const bf16*)Ain;
  const int lr = t >> 3;                 // 0..31 (32 rows per pass)
  const int lc = (t & 7) * 4;            // 4 elements per thread

  for (int k0 = 0; k0 < K; k0 += 32) {
    __syncthreads();
#pragma unroll
    for (int p = 0; p < 4; ++p) {        // A tile (convert fp32->bf16 if needed)
      int r = p * 32 + lr;
      long long g = (rowBase + r) * (long long)K + k0 + lc;
      if (A_IS_BF16) {
        const bf16* s = Ab + g;
        lA[r][lc] = s[0]; lA[r][lc + 1] = s[1]; lA[r][lc + 2] = s[2]; lA[r][lc + 3] = s[3];
      } else {
        const float* s = Af + g;
        lA[r][lc] = (bf16)s[0]; lA[r][lc + 1] = (bf16)s[1];
        lA[r][lc + 2] = (bf16)s[2]; lA[r][lc + 3] = (bf16)s[3];
      }
    }
#pragma unroll
    for (int p = 0; p < 4; ++p) {        // B tile = W rows
      int n = p * 32 + lr;
      const float* s = W + (long long)(colBase + n) * K + k0 + lc;
      lB[n][lc] = (bf16)s[0]; lB[n][lc + 1] = (bf16)s[1];
      lB[n][lc + 2] = (bf16)s[2]; lB[n][lc + 3] = (bf16)s[3];
    }
    __syncthreads();
    if (k0 + 32 < K) {                   // CDNA5 global_prefetch_b8 of next tile
      long long g = (rowBase + lr) * (long long)K + k0 + 32 + lc;
      __builtin_prefetch(A_IS_BF16 ? (const void*)(Ab + g) : (const void*)(Af + g), 0, 1);
    }
    v16bf bfrag[4];
#pragma unroll
    for (int ni = 0; ni < 4; ++ni)
      bfrag[ni] = lds_b_frag(&lB[wn * 64 + ni * 16 + nl][0], half);
#pragma unroll
    for (int mi = 0; mi < 2; ++mi) {
      v16bf a = lds_a_frag(&lA[wm * 32 + mi * 16 + nl][0], half);
#pragma unroll
      for (int ni = 0; ni < 4; ++ni) acc[mi][ni] = WMMA_BF16(a, bfrag[ni], acc[mi][ni]);
    }
  }

  float* Cf = (float*)Cout;
  bf16*  Cb = (bf16*)Cout;
#pragma unroll
  for (int mi = 0; mi < 2; ++mi)
#pragma unroll
    for (int ni = 0; ni < 4; ++ni) {
      int col = colBase + wn * 64 + ni * 16 + nl;
      float bv = bias[col];
#pragma unroll
      for (int r = 0; r < 8; ++r) {
        long long row = rowBase + wm * 32 + mi * 16 + r + 8 * half;
        float v = acc[mi][ni][r] + bv;
        if (OUT_BF16) Cb[row * DO + col] = (bf16)v;
        else          Cf[row * DO + col] = v;
      }
    }
}

// ---------------------------------------------------------------------------
// Kernel 2: phi = exp(Qh @ P - 0.5*||Qh||^2).  Per (b,h): [S,64]x[64,256].
// grid = (M/128, S/128, B*H).  P tile transposed into [N][K] LDS layout.
// ---------------------------------------------------------------------------
__global__ __launch_bounds__(256)
void feature_map_kernel(const bf16* __restrict__ Qkv, const float* __restrict__ P,
                        bf16* __restrict__ Phi, int S, int H, int DH, int M) {
  __shared__ alignas(16) bf16 lA[128][72];
  __shared__ alignas(16) bf16 lB[128][72];
  __shared__ float lnorm[128];
  const int t = threadIdx.x, lane = t & 31, wv = t >> 5;
  const int wm = wv >> 1, wn = wv & 1;
  const int half = lane >> 4, nl = lane & 15;
  const int bh = blockIdx.z, b = bh / H, h = bh % H;
  const long long s0 = (long long)blockIdx.y * 128;
  const int n0 = blockIdx.x * 128;
  const int D = H * DH;

  {                                      // Q tile: [128 rows][64 cols] bf16
    const int lr = t >> 3, lc = (t & 7) * 8;
    for (int p = 0; p < 4; ++p) {
      int r = p * 32 + lr;
      const bf16* s = Qkv + ((long long)b * S + s0 + r) * D + h * DH + lc;
#pragma unroll
      for (int j = 0; j < 8; ++j) lA[r][lc + j] = s[j];
    }
  }
  for (int e = t; e < 64 * 128; e += 256) {  // P transposed: lB[n][k]=P[k][n0+n]
    int k = e >> 7, n = e & 127;
    lB[n][k] = (bf16)P[k * M + n0 + n];
  }
  __syncthreads();
  if (t < 128) {                         // 0.5*||q||^2 per row
    float s = 0.f;
#pragma unroll
    for (int j = 0; j < 64; ++j) { float q = (float)lA[t][j]; s += q * q; }
    lnorm[t] = 0.5f * s;
  }
  __syncthreads();

  v8f zero = {};
  v8f acc[2][4];
  for (int i = 0; i < 2; ++i)
    for (int j = 0; j < 4; ++j) acc[i][j] = zero;
#pragma unroll
  for (int ks = 0; ks < 2; ++ks) {
    int k0 = ks * 32;
    v16bf bfrag[4];
#pragma unroll
    for (int ni = 0; ni < 4; ++ni)
      bfrag[ni] = lds_b_frag(&lB[wn * 64 + ni * 16 + nl][k0], half);
#pragma unroll
    for (int mi = 0; mi < 2; ++mi) {
      v16bf a = lds_a_frag(&lA[wm * 32 + mi * 16 + nl][k0], half);
#pragma unroll
      for (int ni = 0; ni < 4; ++ni) acc[mi][ni] = WMMA_BF16(a, bfrag[ni], acc[mi][ni]);
    }
  }
#pragma unroll
  for (int mi = 0; mi < 2; ++mi)
#pragma unroll
    for (int ni = 0; ni < 4; ++ni) {
      int col = n0 + wn * 64 + ni * 16 + nl;
#pragma unroll
      for (int r = 0; r < 8; ++r) {
        int rl = wm * 32 + mi * 16 + r + 8 * half;
        float v = __expf(acc[mi][ni][r] - lnorm[rl]);
        Phi[((long long)bh * S + s0 + rl) * M + col] = (bf16)v;
      }
    }
}

// ---------------------------------------------------------------------------
// Kernel 3: Z[b,h,m] = sum_s phiK -> column 64 of augmented KVZ (cols 65..79=0)
// ---------------------------------------------------------------------------
__global__ __launch_bounds__(256)
void z_reduce_kernel(const bf16* __restrict__ Kp, float* __restrict__ KVZf,
                     int S, int M) {
  const int bh = blockIdx.x, m = threadIdx.x;  // blockDim == M == 256
  const bf16* src = Kp + (long long)bh * S * M + m;
  float acc = 0.f;
  for (int s = 0; s < S; ++s) acc += (float)src[(long long)s * M];
  float* dst = KVZf + ((long long)bh * M + m) * 80;
  dst[64] = acc;
#pragma unroll
  for (int c = 65; c < 80; ++c) dst[c] = 0.f;
}

// ---------------------------------------------------------------------------
// Kernel 4: KV[m,d] = sum_s phiK[s,m]*V[s,d].  A=phiK^T via LDS transpose,
// split-K over S with global_atomic_add_f32.  grid=(M/128, SPLITK, B*H).
// ---------------------------------------------------------------------------
__global__ __launch_bounds__(256)
void kv_kernel(const bf16* __restrict__ Kp, const bf16* __restrict__ Vb,
               float* __restrict__ KVZf, int S, int H, int DH, int M, int SPLITK) {
  __shared__ alignas(16) bf16 lA[128][40];   // [m][s]
  __shared__ alignas(16) bf16 lB[64][40];    // [d][s]
  const int t = threadIdx.x, lane = t & 31, wv = t >> 5;
  const int half = lane >> 4, nl = lane & 15;
  const int bh = blockIdx.z, b = bh / H, h = bh % H;
  const int m0 = blockIdx.x * 128;
  const int chunk = S / SPLITK;
  const long long sBase = (long long)blockIdx.y * chunk;
  const int D = H * DH;

  v8f zero = {};
  v8f acc[4];
  for (int j = 0; j < 4; ++j) acc[j] = zero;

  for (int s0 = 0; s0 < chunk; s0 += 32) {
    __syncthreads();
    {                                    // phiK transposed -> lA[m][s]
      int m2 = (t & 63) * 2, sr = t >> 6;
      for (int p = 0; p < 8; ++p) {
        int s = p * 4 + sr;
        const bf16* src = Kp + ((long long)bh * S + sBase + s0 + s) * M + m0 + m2;
        lA[m2][s] = src[0]; lA[m2 + 1][s] = src[1];
      }
    }
    {                                    // V transposed -> lB[d][s]
      int d2 = (t & 31) * 2, sr = t >> 5;
      for (int p = 0; p < 4; ++p) {
        int s = p * 8 + sr;
        const bf16* src = Vb + ((long long)b * S + sBase + s0 + s) * D + h * DH + d2;
        lB[d2][s] = src[0]; lB[d2 + 1][s] = src[1];
      }
    }
    __syncthreads();
    v16bf a = lds_a_frag(&lA[wv * 16 + nl][0], half);
#pragma unroll
    for (int ni = 0; ni < 4; ++ni) {
      v16bf bfrag = lds_b_frag(&lB[ni * 16 + nl][0], half);
      acc[ni] = WMMA_BF16(a, bfrag, acc[ni]);
    }
  }
#pragma unroll
  for (int ni = 0; ni < 4; ++ni) {
    int d = ni * 16 + nl;
#pragma unroll
    for (int r = 0; r < 8; ++r) {
      int m = m0 + wv * 16 + r + 8 * half;
      atomicAdd(&KVZf[((long long)bh * M + m) * 80 + d], acc[ni][r]);
    }
  }
}

__global__ __launch_bounds__(256)
void pack_kvz_kernel(const float* __restrict__ KVZf, bf16* __restrict__ KVZb, int n) {
  int i = blockIdx.x * 256 + threadIdx.x;
  if (i < n) KVZb[i] = (bf16)KVZf[i];
}

// ---------------------------------------------------------------------------
// Kernel 5: [S,256] x [256,80] GEMM; N-tile 4 column 64 is the denominator.
// Fused divide + head-interleaved bf16 store.  grid=(S/128, B*H).
// ---------------------------------------------------------------------------
__global__ __launch_bounds__(256)
void numden_kernel(const bf16* __restrict__ Qp, const bf16* __restrict__ KVZb,
                   bf16* __restrict__ Att, int S, int H, int DH, int M) {
  __shared__ alignas(16) bf16 lA[128][40];
  __shared__ alignas(16) bf16 lB[80][40];
  __shared__ float lden[128];
  const int t = threadIdx.x, lane = t & 31, wv = t >> 5;
  const int half = lane >> 4, nl = lane & 15;
  const int bh = blockIdx.y, b = bh / H, h = bh % H;
  const long long s0 = (long long)blockIdx.x * 128;
  const int D = H * DH;

  v8f zero = {};
  v8f acc[5];
  for (int j = 0; j < 5; ++j) acc[j] = zero;

  for (int k0 = 0; k0 < M; k0 += 32) {
    __syncthreads();
    {                                    // phiQ tile row-major
      int lr = t >> 2, lc = (t & 3) * 8;
      for (int p = 0; p < 2; ++p) {
        int r = p * 64 + lr;
        const bf16* s = Qp + ((long long)bh * S + s0 + r) * M + k0 + lc;
#pragma unroll
        for (int j = 0; j < 8; ++j) lA[r][lc + j] = s[j];
      }
    }
    for (int e = t; e < 80 * 32; e += 256) {   // KVZ transposed: lB[n][k]
      int k = e / 80, n = e - k * 80;
      lB[n][k] = KVZb[((long long)bh * M + k0 + k) * 80 + n];
    }
    __syncthreads();
    v16bf a = lds_a_frag(&lA[wv * 16 + nl][0], half);
#pragma unroll
    for (int ni = 0; ni < 5; ++ni) {
      v16bf bfrag = lds_b_frag(&lB[ni * 16 + nl][0], half);
      acc[ni] = WMMA_BF16(a, bfrag, acc[ni]);
    }
  }
  if (nl == 0) {                         // column 64 -> per-row denominator
#pragma unroll
    for (int r = 0; r < 8; ++r) lden[wv * 16 + r + 8 * half] = acc[4][r];
  }
  __syncthreads();
#pragma unroll
  for (int ni = 0; ni < 4; ++ni) {
    int d = ni * 16 + nl;
#pragma unroll
    for (int r = 0; r < 8; ++r) {
      int rl = wv * 16 + r + 8 * half;
      float den = lden[rl];
      den = den > 1e-6f ? den : 1e-6f;
      Att[((long long)b * S + s0 + rl) * D + h * DH + d] = (bf16)(acc[ni][r] / den);
    }
  }
}

// ---------------------------------------------------------------------------
extern "C" void kernel_launch(void* const* d_in, const int* in_sizes, int n_in,
                              void* d_out, int out_size, void* d_ws, size_t ws_size,
                              hipStream_t stream) {
  (void)in_sizes; (void)n_in; (void)out_size; (void)ws_size;
  const float* x  = (const float*)d_in[0];
  const float* Wq = (const float*)d_in[1];
  const float* bq = (const float*)d_in[2];
  const float* Wk = (const float*)d_in[3];
  const float* bk = (const float*)d_in[4];
  const float* Wv = (const float*)d_in[5];
  const float* bv = (const float*)d_in[6];
  const float* Wo = (const float*)d_in[7];
  const float* bo = (const float*)d_in[8];
  const float* P  = (const float*)d_in[9];
  float* out = (float*)d_out;

  const int Bb = 2, S = 8192, D = 1024, H = 16, DH = 64, M = 256;
  const long long N = (long long)Bb * S;   // 16384
  const int BH = Bb * H;                   // 32

  char* ws = (char*)d_ws;
  const size_t qkvBytes = (size_t)N * D * 2;          // 32 MiB each (bf16)
  const size_t phiBytes = (size_t)BH * S * M * 2;     // 128 MiB each (bf16)
  bf16* Qb = (bf16*)(ws);
  bf16* Kb = (bf16*)(ws + qkvBytes);
  bf16* Vb = (bf16*)(ws + 2 * qkvBytes);
  bf16* Qp = (bf16*)(ws + 3 * qkvBytes);
  bf16* Kp = (bf16*)(ws + 3 * qkvBytes + phiBytes);
  const size_t kvzElems = (size_t)BH * M * 80;
  float* KVZf = (float*)(ws + 3 * qkvBytes + 2 * phiBytes);
  bf16*  KVZb = (bf16*)(ws + 3 * qkvBytes + 2 * phiBytes + kvzElems * 4);
  bf16*  Att  = Qb;  // Qb is dead after the Q feature map -> reuse for attention out

  dim3 blk(256);
  dim3 gGemm(D / 128, (unsigned)(N / 128));            // (8, 128)
  gemm_bias_kernel<false, true><<<gGemm, blk, 0, stream>>>(x, Wq, bq, Qb, D, D);
  gemm_bias_kernel<false, true><<<gGemm, blk, 0, stream>>>(x, Wk, bk, Kb, D, D);
  gemm_bias_kernel<false, true><<<gGemm, blk, 0, stream>>>(x, Wv, bv, Vb, D, D);

  dim3 gFeat(M / 128, S / 128, BH);                    // (2, 64, 32)
  feature_map_kernel<<<gFeat, blk, 0, stream>>>(Qb, P, Qp, S, H, DH, M);
  feature_map_kernel<<<gFeat, blk, 0, stream>>>(Kb, P, Kp, S, H, DH, M);

  hipMemsetAsync(KVZf, 0, kvzElems * 4, stream);       // zero for split-K atomics
  z_reduce_kernel<<<dim3(BH), dim3(256), 0, stream>>>(Kp, KVZf, S, M);
  dim3 gKV(M / 128, 16, BH);                           // split-K = 16
  kv_kernel<<<gKV, blk, 0, stream>>>(Kp, Vb, KVZf, S, H, DH, M, 16);
  pack_kvz_kernel<<<dim3((unsigned)((kvzElems + 255) / 256)), blk, 0, stream>>>(
      KVZf, KVZb, (int)kvzElems);

  dim3 gND(S / 128, BH);
  numden_kernel<<<gND, blk, 0, stream>>>(Qp, KVZb, Att, S, H, DH, M);

  gemm_bias_kernel<true, false><<<gGemm, blk, 0, stream>>>(Att, Wo, bo, out, D, D);
}
